// PATCH_Attention_47038481825862
// MI455X (gfx1250) — matrix operand order repaired and checked by reading, not compiled, not run on hardware
//
#include <hip/hip_runtime.h>
#include <hip/hip_bf16.h>
#include <math.h>

// ---------------------------------------------------------------------------
// Problem constants (match reference setup_inputs: B=32, SIDE=32, D=768)
// ---------------------------------------------------------------------------
#define B_    32
#define SIDE_ 32
#define D_    768
#define C_    1024                 // center rows = SIDE*SIDE
#define S_    1088                 // SIDE + C + SIDE
#define LDA_  776                  // padded bf16 row stride in LDS (768 + 8)

typedef __attribute__((ext_vector_type(16))) __bf16 v16bf;
typedef __attribute__((ext_vector_type(8)))  __bf16 v8bf;
typedef __attribute__((ext_vector_type(8)))  float  v8f;

static __device__ inline unsigned pack_bf16x2(float a, float b) {
  union { __bf16 h[2]; unsigned u; } p;
  p.h[0] = (__bf16)a;
  p.h[1] = (__bf16)b;
  return p.u;
}

static __device__ inline v16bf cvt16(float4 f0, float4 f1, float4 f2, float4 f3) {
  union { v16bf v; __bf16 e[16]; } u;
  u.e[0]  = (__bf16)f0.x; u.e[1]  = (__bf16)f0.y;
  u.e[2]  = (__bf16)f0.z; u.e[3]  = (__bf16)f0.w;
  u.e[4]  = (__bf16)f1.x; u.e[5]  = (__bf16)f1.y;
  u.e[6]  = (__bf16)f1.z; u.e[7]  = (__bf16)f1.w;
  u.e[8]  = (__bf16)f2.x; u.e[9]  = (__bf16)f2.y;
  u.e[10] = (__bf16)f2.z; u.e[11] = (__bf16)f2.w;
  u.e[12] = (__bf16)f3.x; u.e[13] = (__bf16)f3.y;
  u.e[14] = (__bf16)f3.z; u.e[15] = (__bf16)f3.w;
  return u.v;
}

// ---------------------------------------------------------------------------
// K1: per c-row inverse L2 norm  +  copy c verbatim into the output middle.
// One wave (32 lanes) per row; 4096 blocks x 256 threads (8 waves/block).
// ---------------------------------------------------------------------------
__global__ __launch_bounds__(256)
void k_norm_copy(const float* __restrict__ x, float* __restrict__ out,
                 float* __restrict__ invc) {
  const int wave = __builtin_amdgcn_readfirstlane(threadIdx.x >> 5);
  const int lane = threadIdx.x & 31;
  const int g    = blockIdx.x * 8 + wave;     // row id in [0, B*C)
  const int b    = g >> 10;
  const int j    = g & 1023;
  const size_t row = (size_t)b * S_ + SIDE_ + j;   // same row index in x and out

  const float4* __restrict__ src = (const float4*)(x   + row * D_);
  float4*       __restrict__ dst = (float4*)      (out + row * D_);

  float  ss = 0.f;
  float4 v[6];
#pragma unroll
  for (int q = 0; q < 6; ++q) {               // 6 * 32 lanes * 4 = 768 floats
    v[q] = src[lane + 32 * q];
    ss += v[q].x * v[q].x + v[q].y * v[q].y + v[q].z * v[q].z + v[q].w * v[q].w;
  }
#pragma unroll
  for (int off = 16; off > 0; off >>= 1) ss += __shfl_xor(ss, off, 32);

#pragma unroll
  for (int q = 0; q < 6; ++q) dst[lane + 32 * q] = v[q];   // fused middle copy

  if (lane == 0) invc[g] = 1.0f / fmaxf(sqrtf(ss), 1e-12f);
}

// ---------------------------------------------------------------------------
// K2: dots (bf16 WMMA, fp32 acc) -> argmax over j -> gather rows to out edges.
// Grid = B*4 blocks; block = 256 threads = 8 waves; each block owns a 16-row
// s-tile. Each wave processes PAIRS of adjacent 16-wide j-tiles (2 independent
// WMMA accumulate chains sharing one A fragment), 4 pair-iterations per wave.
// ---------------------------------------------------------------------------
__global__ __launch_bounds__(256)
void k_attn(const float* __restrict__ x, const float* __restrict__ invc,
            float* __restrict__ out) {
  __shared__ __attribute__((aligned(16))) __bf16 sA[16 * LDA_];  // 24.25 KB
  __shared__ float redv[8][16];
  __shared__ int   redj[8][16];
  __shared__ int   idxs[16];

  const int b    = blockIdx.x >> 2;
  const int t    = blockIdx.x & 3;           // which 16-row s-tile (0..3)
  const int tid  = threadIdx.x;
  const int lane = tid & 31;
  const int wave = __builtin_amdgcn_readfirstlane(tid >> 5);  // scalar -> SGPR loop

  // ---- stage the 16 s-rows as bf16 into LDS (uint2 = 4 packed bf16) -------
#pragma unroll
  for (int it = 0; it < 12; ++it) {          // 12 * 256 = 16 rows * 192 float4
    const int e  = tid + 256 * it;
    const int m  = e / 192;                  // row in tile
    const int q  = e % 192;                  // float4 index within the row
    const int i  = t * 16 + m;               // s-row index 0..63
    const int xr = (i < SIDE_) ? i : (C_ + i);   // l rows : r rows in x
    const float4 f = *(const float4*)(x + ((size_t)b * S_ + xr) * D_ + q * 4);
    uint2 u;
    u.x = pack_bf16x2(f.x, f.y);
    u.y = pack_bf16x2(f.z, f.w);
    *(uint2*)(&sA[m * LDA_ + q * 4]) = u;
  }
  __syncthreads();

  // ---- WMMA fragment roles (per CDNA5 ISA layouts) --------------
  const int n  = lane & 15;                  // B/D column within 16-wide tile
  const int kh = lane >> 4;                  // K-half selector (A/B), row-half (D)
  const __bf16* __restrict__ arow = &sA[(lane & 15) * LDA_];

  float bestv[8];
  int   bestj[8];
#pragma unroll
  for (int r = 0; r < 8; ++r) { bestv[r] = -INFINITY; bestj[r] = 0; }

  const float* __restrict__ cbase = x + ((size_t)b * S_ + SIDE_) * D_;
  const float* __restrict__ invb  = invc + b * C_;

  for (int p = 0; p < 4; ++p) {              // scalar loop (wave is SGPR)
    const int jt0    = 2 * wave + 16 * p;    // this wave's j-tile pair
    const int jbase0 = jt0 * 16;
    const int jbase1 = jbase0 + 16;

    // hoist the per-column norm scales above the k-loop (latency hidden)
    const float inv0 = invb[jbase0 + n];
    const float inv1 = invb[jbase1 + n];

    const float* __restrict__ crow0 = cbase + (size_t)(jbase0 + n) * D_ + kh * 16;
    const float* __restrict__ crow1 = crow0 + (size_t)16 * D_;

    v8f acc0 = {};
    v8f acc1 = {};
    for (int k = 0; k < D_; k += 32) {
      // B fragments: 16 consecutive K elements of c rows (jbase+n), K-half kh
      const float4* cp0 = (const float4*)(crow0 + k);
      const float4* cp1 = (const float4*)(crow1 + k);
      const v16bf b0 = cvt16(cp0[0], cp0[1], cp0[2], cp0[3]);
      const v16bf b1 = cvt16(cp1[0], cp1[1], cp1[2], cp1[3]);

      // A fragment from LDS: row (lane&15); K = k + kh*16 + [0..15]
      union { v16bf v; v8bf h[2]; } au;
      au.h[0] = *(const v8bf*)(arow + k + kh * 16);
      au.h[1] = *(const v8bf*)(arow + k + kh * 16 + 8);

      // two independent accumulate chains sharing one A fragment
      acc0 = __builtin_amdgcn_wmma_f32_16x16x32_bf16(
          false, au.v, false, b0, (short)0, acc0, false, false);
      acc1 = __builtin_amdgcn_wmma_f32_16x16x32_bf16(
          false, au.v, false, b1, (short)0, acc1, false, false);
    }

    // scale by 1/||c_j|| (fp32) and fold into the running per-row argmax.
    // tile jbase0 first, then jbase1 (ascending j) -> '>' keeps lowest j.
    const int jcol0 = jbase0 + n;
    const int jcol1 = jbase1 + n;
#pragma unroll
    for (int r = 0; r < 8; ++r) {            // rows m = kh*8 + r
      const float v0 = acc0[r] * inv0;
      if (v0 > bestv[r]) { bestv[r] = v0; bestj[r] = jcol0; }
      const float v1 = acc1[r] * inv1;
      if (v1 > bestv[r]) { bestv[r] = v1; bestj[r] = jcol1; }
    }
  }

  // ---- argmax reduction across the 16 columns (within each 16-lane half) --
#pragma unroll
  for (int r = 0; r < 8; ++r) {
#pragma unroll
    for (int off = 1; off < 16; off <<= 1) {
      const float ov = __shfl_xor(bestv[r], off, 32);
      const int   oj = __shfl_xor(bestj[r], off, 32);
      if (ov > bestv[r] || (ov == bestv[r] && oj < bestj[r])) {
        bestv[r] = ov; bestj[r] = oj;
      }
    }
  }
  if ((lane & 15) == 0) {                    // lane 0 -> rows 0..7, lane 16 -> 8..15
#pragma unroll
    for (int r = 0; r < 8; ++r) {
      redv[wave][kh * 8 + r] = bestv[r];
      redj[wave][kh * 8 + r] = bestj[r];
    }
  }
  __syncthreads();

  // ---- cross-wave reduction (8 candidates per row, tie -> lowest j) -------
  if (tid < 16) {
    float bv = redv[0][tid];
    int   bj = redj[0][tid];
#pragma unroll
    for (int w = 1; w < 8; ++w) {
      const float ov = redv[w][tid];
      const int   oj = redj[w][tid];
      if (ov > bv || (ov == bv && oj < bj)) { bv = ov; bj = oj; }
    }
    idxs[tid] = bj;
  }
  __syncthreads();

  // ---- gather: copy the 16 selected fp32 c rows to the output edges -------
#pragma unroll
  for (int it = 0; it < 12; ++it) {
    const int e  = tid + 256 * it;
    const int m  = e / 192;
    const int q  = e % 192;
    const int i  = t * 16 + m;
    const int dr = (i < SIDE_) ? i : (C_ + i);     // 1056 + (i-32) == 1024 + i
    const float4 f =
        *(const float4*)(x + ((size_t)b * S_ + SIDE_ + idxs[m]) * D_ + q * 4);
    *(float4*)(out + ((size_t)b * S_ + dr) * D_ + q * 4) = f;
  }
}

// ---------------------------------------------------------------------------
// Launch: K1 (norms + middle copy) then K2 (WMMA dots + argmax + gather).
// ws usage: B*C floats of inv_cnorm = 128 KB.
// ---------------------------------------------------------------------------
extern "C" void kernel_launch(void* const* d_in, const int* in_sizes, int n_in,
                              void* d_out, int out_size, void* d_ws, size_t ws_size,
                              hipStream_t stream) {
  (void)in_sizes; (void)n_in; (void)out_size; (void)ws_size;
  const float* x    = (const float*)d_in[0];   // d_in[1] = side (==32), hardcoded
  float*       out  = (float*)d_out;
  float*       invc = (float*)d_ws;            // B_*C_ floats

  k_norm_copy<<<dim3((B_ * C_) / 8), dim3(256), 0, stream>>>(x, out, invc);
  k_attn<<<dim3(B_ * 4), dim3(256), 0, stream>>>(x, invc, out);
}
